// CSTreeLSTMCell_7456063226111
// MI455X (gfx1250) — matrix operand order repaired and checked by reading, not compiled, not run on hardware
//
#include <hip/hip_runtime.h>
#include <hip/hip_bf16.h>

#define BATCH 32768
#define D 128
#define NCH 8

typedef __bf16 bf16;
typedef __attribute__((ext_vector_type(16))) __bf16 v16bf;
typedef __attribute__((ext_vector_type(8)))  __bf16 v8bf;
typedef __attribute__((ext_vector_type(8)))  float  v8f;

// ---------- helpers ----------

__device__ __forceinline__ unsigned short bfbits(float f) {
  // round-to-nearest-even f32 -> bf16 bit pattern
  unsigned u = __builtin_bit_cast(unsigned, f);
  unsigned r = u + 0x7FFFu + ((u >> 16) & 1u);
  return (unsigned short)(r >> 16);
}

__device__ __forceinline__ unsigned pack2(float a, float b) {
  return (unsigned)bfbits(a) | ((unsigned)bfbits(b) << 16);
}

__device__ __forceinline__ float sigmoidf_fast(float z) {
  return 1.0f / (1.0f + __expf(-z));
}

__device__ __forceinline__ float tanhf_fast(float z) {
  z = fminf(fmaxf(z, -15.0f), 15.0f);
  float e = __expf(2.0f * z);
  return (e - 1.0f) / (e + 1.0f);
}

// A-fragment (16x32 bf16, MxK) from bf16 row-major [rows, D].
// Lanes 0-15: row = lane, elems 0..7 = K kk+0..7, elems 8..15 = K kk+16..23.
// Lanes 16-31: row = lane-16, K halves shifted by +8 (kA).
// Two contiguous 16B loads; shuffle is register renaming only.
__device__ __forceinline__ v16bf load_a_bf(const bf16* m, int row, int kk, int kA) {
  const bf16* p = m + (size_t)row * D + kk + kA;
  v8bf lo = *(const v8bf*)p;
  v8bf hi = *(const v8bf*)(p + 16);
  return __builtin_shufflevector(lo, hi, 0, 1, 2, 3, 4, 5, 6, 7,
                                 8, 9, 10, 11, 12, 13, 14, 15);
}

// B-fragment (32x16 bf16, KxN) from pre-transposed bf16 weights Wt[N=128][K=128].
// Lanes 0-15: col n = lane, K = kk+0..7,16..23 ; lanes 16-31: col n = lane-16, K = kk+8..15,24..31.
__device__ __forceinline__ v16bf load_b_t(const bf16* wt, int nB, int kk, int kB) {
  const bf16* p = wt + (size_t)nB * D + kk + kB;
  v8bf lo = *(const v8bf*)p;
  v8bf hi = *(const v8bf*)(p + 16);
  return __builtin_shufflevector(lo, hi, 0, 1, 2, 3, 4, 5, 6, 7,
                                 8, 9, 10, 11, 12, 13, 14, 15);
}

__device__ __forceinline__ v8f wmma_bf16(v16bf a, v16bf b, v8f c) {
  return __builtin_amdgcn_wmma_f32_16x16x32_bf16(false, a, false, b, (short)0, c,
                                                 false, false);
}

// ---------- prep kernels (bandwidth-bound; all cvt VALU lives here) ----------

// Read h once: emit bf16 copies of each child and bf16 child-sum.
__global__ void treelstm_prep_h(const float* __restrict__ h,
                                uint2* __restrict__ hb,    // 8 * B*D/4
                                uint2* __restrict__ htb) { // B*D/4
  size_t i = (size_t)blockIdx.x * blockDim.x + threadIdx.x;  // one float4 each
  const float4* h4 = (const float4*)h;
  float4 s = make_float4(0.f, 0.f, 0.f, 0.f);
#pragma unroll
  for (int n = 0; n < NCH; ++n) {
    float4 v = h4[(size_t)n * (BATCH * D / 4) + i];
    hb[(size_t)n * (BATCH * D / 4) + i] =
        make_uint2(pack2(v.x, v.y), pack2(v.z, v.w));
    s.x += v.x; s.y += v.y; s.z += v.z; s.w += v.w;
  }
  htb[i] = make_uint2(pack2(s.x, s.y), pack2(s.z, s.w));
}

__global__ void treelstm_prep_x(const float* __restrict__ x,
                                uint2* __restrict__ xb) {
  size_t i = (size_t)blockIdx.x * blockDim.x + threadIdx.x;
  float4 v = ((const float4*)x)[i];
  xb[i] = make_uint2(pack2(v.x, v.y), pack2(v.z, v.w));
}

// Wt[n][k] = bf16(W[k][n]) with packed 2-element stores along k.
__global__ void treelstm_wtrans(const float* __restrict__ w,
                                unsigned* __restrict__ wt) {
  int i = blockIdx.x * blockDim.x + threadIdx.x;  // 0..8191
  int n = i >> 6;
  int k = (i & 63) * 2;
  wt[i] = pack2(w[(size_t)k * D + n], w[(size_t)(k + 1) * D + n]);
}

// ---------- fused main kernel ----------
// grid.x = BATCH/16 row tiles; 256 threads = 8 waves; wave w -> output cols 16w..16w+15.
__global__ __launch_bounds__(256) void treelstm_main(
    const bf16* __restrict__ xb, const bf16* __restrict__ hb,
    const bf16* __restrict__ htb, const float* __restrict__ C,
    const bf16* __restrict__ WtAll,  // 8 transposed bf16 mats: Wi,Ui,Wo,Uo,Wu,Uu,Wf,Uf
    const float* __restrict__ b_i, const float* __restrict__ b_f,
    const float* __restrict__ b_o, const float* __restrict__ b_u,
    float* __restrict__ h_out, float* __restrict__ c_out) {
  const int lane = threadIdx.x & 31;
  const int wave = threadIdx.x >> 5;
  const int row0 = blockIdx.x * 16;
  const int n0 = wave * 16;

  const int mA = lane & 15;                   // A-frag row in tile
  const int kA = (lane >> 4) << 3;            // 0 or 8
  const int nB = n0 + (lane & 15);            // B-frag global column
  const int kB = (lane >> 4) << 3;            // 0 or 8
  const int nD = n0 + (lane & 15);            // C/D-frag column
  const int mD0 = row0 + ((lane >> 4) << 3);  // C/D-frag base row (+v)

  const bf16* WiT = WtAll + 0 * (size_t)D * D;
  const bf16* UiT = WtAll + 1 * (size_t)D * D;
  const bf16* WoT = WtAll + 2 * (size_t)D * D;
  const bf16* UoT = WtAll + 3 * (size_t)D * D;
  const bf16* WuT = WtAll + 4 * (size_t)D * D;
  const bf16* UuT = WtAll + 5 * (size_t)D * D;
  const bf16* WfT = WtAll + 6 * (size_t)D * D;
  const bf16* UfT = WtAll + 7 * (size_t)D * D;

  v8f acc_i = {}, acc_o = {}, acc_u = {}, acc_fx = {};

  // gate GEMMs: x@W_* and h_tilde@U_* over K = 128
#pragma unroll
  for (int kk = 0; kk < D; kk += 32) {
    v16bf ax = load_a_bf(xb, row0 + mA, kk, kA);
    v16bf ah = load_a_bf(htb, row0 + mA, kk, kA);
    acc_i  = wmma_bf16(ax, load_b_t(WiT, nB, kk, kB), acc_i);
    acc_i  = wmma_bf16(ah, load_b_t(UiT, nB, kk, kB), acc_i);
    acc_o  = wmma_bf16(ax, load_b_t(WoT, nB, kk, kB), acc_o);
    acc_o  = wmma_bf16(ah, load_b_t(UoT, nB, kk, kB), acc_o);
    acc_u  = wmma_bf16(ax, load_b_t(WuT, nB, kk, kB), acc_u);
    acc_u  = wmma_bf16(ah, load_b_t(UuT, nB, kk, kB), acc_u);
    acc_fx = wmma_bf16(ax, load_b_t(WfT, nB, kk, kB), acc_fx);
  }

  // preload U_f B-fragments once; reused by all 8 children
  v16bf bUf[4];
#pragma unroll
  for (int t = 0; t < 4; ++t) bUf[t] = load_b_t(UfT, nB, t * 32, kB);

  const float bi = b_i[nD], bo = b_o[nD], bu = b_u[nD], bf = b_f[nD];

  v8f csum;
#pragma unroll
  for (int v = 0; v < 8; ++v) {
    float iv = sigmoidf_fast(acc_i[v] + bi);
    float ov = sigmoidf_fast(acc_o[v] + bo);
    float uv = tanhf_fast(acc_u[v] + bu);
    acc_o[v] = ov;      // keep o for the output stage
    csum[v] = iv * uv;  // i * u
  }

  // per-child: f_n = sigmoid(x@W_f + h_n@U_f + b_f); csum += f_n * C_n
  for (int n = 0; n < NCH; ++n) {
    if (n + 1 < NCH)
      __builtin_prefetch(hb + ((size_t)(n + 1) * BATCH + row0 + mA) * D, 0, 1);
    const bf16* hn = hb + (size_t)n * BATCH * D;
    v8f accf = acc_fx;
#pragma unroll
    for (int t = 0; t < 4; ++t) {
      v16bf ahn = load_a_bf(hn, row0 + mA, t * 32, kA);
      accf = wmma_bf16(ahn, bUf[t], accf);
    }
    const float* Cn = C + (size_t)n * BATCH * D;
#pragma unroll
    for (int v = 0; v < 8; ++v) {
      float f = sigmoidf_fast(accf[v] + bf);
      csum[v] += f * Cn[(size_t)(mD0 + v) * D + nD];
    }
  }

  // c = i*u + sum f*C ; h = o * tanh(c)
#pragma unroll
  for (int v = 0; v < 8; ++v) {
    float c = csum[v];
    size_t idx = (size_t)(mD0 + v) * D + nD;
    c_out[idx] = c;
    h_out[idx] = acc_o[v] * tanhf_fast(c);
  }
}

// ---------- launch ----------

extern "C" void kernel_launch(void* const* d_in, const int* in_sizes, int n_in,
                              void* d_out, int out_size, void* d_ws, size_t ws_size,
                              hipStream_t stream) {
  const float* x   = (const float*)d_in[0];
  const float* h   = (const float*)d_in[1];
  const float* C   = (const float*)d_in[2];
  const float* W_i = (const float*)d_in[3];
  const float* W_f = (const float*)d_in[4];
  const float* W_o = (const float*)d_in[5];
  const float* W_u = (const float*)d_in[6];
  const float* U_i = (const float*)d_in[7];
  const float* U_f = (const float*)d_in[8];
  const float* U_o = (const float*)d_in[9];
  const float* U_u = (const float*)d_in[10];
  const float* b_i = (const float*)d_in[11];
  const float* b_f = (const float*)d_in[12];
  const float* b_o = (const float*)d_in[13];
  const float* b_u = (const float*)d_in[14];

  const size_t elems = (size_t)BATCH * D;  // 4,194,304
  char* ws = (char*)d_ws;
  bf16* xb    = (bf16*)ws;                    ws += elems * sizeof(bf16);        // 8.4 MB
  bf16* htb   = (bf16*)ws;                    ws += elems * sizeof(bf16);        // 8.4 MB
  bf16* hbAll = (bf16*)ws;                    ws += NCH * elems * sizeof(bf16);  // 67 MB
  bf16* WtAll = (bf16*)ws;                                                      // 256 KB

  treelstm_prep_x<<<(int)(elems / 4 / 256), 256, 0, stream>>>(x, (uint2*)xb);
  treelstm_prep_h<<<(int)(elems / 4 / 256), 256, 0, stream>>>(h, (uint2*)hbAll,
                                                              (uint2*)htb);

  const float* srcs[8] = {W_i, U_i, W_o, U_o, W_u, U_u, W_f, U_f};
  for (int m = 0; m < 8; ++m)
    treelstm_wtrans<<<(D * D / 2) / 256, 256, 0, stream>>>(
        srcs[m], (unsigned*)(WtAll + (size_t)m * D * D));

  float* h_out = (float*)d_out;
  float* c_out = h_out + elems;
  treelstm_main<<<BATCH / 16, 256, 0, stream>>>(xb, hbAll, htb, C, WtAll, b_i, b_f,
                                                b_o, b_u, h_out, c_out);
}
